// GATv2Layer_25744033972329
// MI455X (gfx1250) — compile-verified, matrix operand order checked by reading, not run on hardware
//
#include <hip/hip_runtime.h>

typedef __attribute__((ext_vector_type(16))) _Float16 v16h;
typedef __attribute__((ext_vector_type(8)))  _Float16 v8h;
typedef __attribute__((ext_vector_type(8)))  float    v8f;

#define IN_CH 128
#define HC    128   // HEADS * OUT_CH
#define HEADS 4
#define GEMM_WAVES 8

// ---------- helpers: order-preserving float<->uint for atomic max ----------
__device__ __forceinline__ unsigned f2ord(float f) {
  unsigned u = __float_as_uint(f);
  return (u & 0x80000000u) ? ~u : (u | 0x80000000u);
}
__device__ __forceinline__ float ord2f(unsigned u) {
  unsigned v = (u & 0x80000000u) ? (u & 0x7fffffffu) : ~u;
  return __uint_as_float(v);
}

// ---------- init: zero accumulators (harness poisons buffers with 0xAA) ----
__global__ void k_init(float* __restrict__ out, unsigned* __restrict__ maxb,
                       float* __restrict__ denom, float* __restrict__ sums,
                       int nOut, int nMax, int nSums) {
  int i = blockIdx.x * blockDim.x + threadIdx.x;
  int stride = gridDim.x * blockDim.x;
  for (int t = i; t < nOut; t += stride) out[t] = 0.f;
  for (int t = i; t < nMax; t += stride) { maxb[t] = 0u; denom[t] = 0.f; }
  for (int t = i; t < nSums; t += stride) sums[t] = 0.f;
}

// ---------- convert x -> f16 ----------
__global__ void k_cvt_x(const float* __restrict__ x, _Float16* __restrict__ xh, int n) {
  int i = blockIdx.x * blockDim.x + threadIdx.x;
  if (i < n) xh[i] = (_Float16)x[i];
}

// ---------- convert + transpose both weights -> f16, wt[m][n][k] = W_m[k][n]
__global__ void k_cvt_w(const float* __restrict__ Ws, const float* __restrict__ Wd,
                        _Float16* __restrict__ wt) {
  int i = blockIdx.x * blockDim.x + threadIdx.x;
  if (i >= 2 * IN_CH * HC) return;
  int m = i / (IN_CH * HC);
  int k = (i / HC) % IN_CH;
  int n = i % HC;
  const float* W = m ? Wd : Ws;
  wt[(size_t)m * IN_CH * HC + (size_t)n * IN_CH + k] = (_Float16)W[(size_t)k * HC + n];
}

// ---------- fused dual projection GEMM via v_wmma_f32_16x16x32_f16 ----------
// 8 waves/block, one 16-row tile per wave. Weights (both matrices, 64 KB f16)
// are staged once per block into LDS with a padded row stride (272 B) so the
// 16-lane ds_load_b128 fragment reads spread across all 64 banks. All four B
// fragments are preloaded per n-tile so the 4 WMMAs issue back-to-back.
__global__ void __launch_bounds__(256) k_gemm(const _Float16* __restrict__ xh,
                                              const _Float16* __restrict__ wt,
                                              float* __restrict__ h_src,
                                              float* __restrict__ h_dst, int N) {
  constexpr int WPAD = 136;                 // 128 + 8 halves -> 272 B row stride
  __shared__ _Float16 sw[2 * 128 * WPAD];   // 69,632 B of the 320 KB LDS

  // cooperative stage: 4096 x 16 B chunks
  {
    const v8h* src = (const v8h*)wt;
    for (int c = threadIdx.x; c < 2 * 128 * 16; c += 256) {
      int row  = c >> 4;         // mat*128 + n
      int colh = (c & 15) * 8;   // half index within row
      *(v8h*)(sw + row * WPAD + colh) = src[c];
    }
  }
  __syncthreads();

  const int lane = threadIdx.x & 31;
  const int wave = threadIdx.x >> 5;
  const int m    = lane & 15;    // A row / B col / C col
  const int hlf  = lane >> 4;    // K sub-block selector
  const int tile = blockIdx.x * GEMM_WAVES + wave;
  if (tile * 16 >= N) return;
  const int  row  = tile * 16 + m;
  const bool full = (tile * 16 + 16) <= N;   // wave-uniform

  // A fragments: 16x32 f16 per ktile; lane holds two contiguous 16 B chunks.
  v16h a[4];
  if (full) {
    const _Float16* xr = xh + (size_t)row * IN_CH + hlf * 8;
#pragma unroll
    for (int kt = 0; kt < 4; ++kt) {
      ((v8h*)&a[kt])[0] = *(const v8h*)(xr + kt * 32);
      ((v8h*)&a[kt])[1] = *(const v8h*)(xr + kt * 32 + 16);
    }
  } else {
#pragma unroll
    for (int kt = 0; kt < 4; ++kt) {
      v16h t = {};
      if (row < N) {
        const _Float16* p = xh + (size_t)row * IN_CH + kt * 32 + hlf * 8;
        ((v8h*)&t)[0] = *(const v8h*)(p);
        ((v8h*)&t)[1] = *(const v8h*)(p + 16);
      }
      a[kt] = t;
    }
  }

#pragma unroll
  for (int mat = 0; mat < 2; ++mat) {
    float* __restrict__ H = mat ? h_dst : h_src;
    const _Float16* sb = sw + mat * 128 * WPAD;
#pragma unroll
    for (int nt = 0; nt < 8; ++nt) {
      // preload all 4 B fragments from LDS
      v16h b[4];
      const _Float16* wr = sb + (nt * 16 + m) * WPAD + hlf * 8;
#pragma unroll
      for (int kt = 0; kt < 4; ++kt) {
        ((v8h*)&b[kt])[0] = *(const v8h*)(wr + kt * 32);
        ((v8h*)&b[kt])[1] = *(const v8h*)(wr + kt * 32 + 16);
      }
      v8f acc = {};
#pragma unroll
      for (int kt = 0; kt < 4; ++kt) {
        acc = __builtin_amdgcn_wmma_f32_16x16x32_f16(
            /*neg_a=*/false, a[kt], /*neg_b=*/false, b[kt],
            /*c_mod=*/(short)0, acc, /*reuse_a=*/false, /*reuse_b=*/false);
      }
      // C/D layout: reg r -> row (r + 8*hlf), col m
      float* Hb = H + (size_t)(tile * 16 + hlf * 8) * HC + nt * 16 + m;
      if (full) {
#pragma unroll
        for (int r = 0; r < 8; ++r) Hb[(size_t)r * HC] = acc[r];
      } else {
#pragma unroll
        for (int r = 0; r < 8; ++r)
          if (tile * 16 + hlf * 8 + r < N) Hb[(size_t)r * HC] = acc[r];
      }
    }
  }
}

// ---------- E1: edge logits + segment max (one wave per edge) ----------
__global__ void __launch_bounds__(256) k_edge_logits(
    const int* __restrict__ ei, int E, int N,
    const float* __restrict__ h_src, const float* __restrict__ h_dst,
    const float* __restrict__ att,
    float* __restrict__ logits, unsigned* __restrict__ maxb) {
  int wid  = (blockIdx.x * blockDim.x + threadIdx.x) >> 5;
  int lane = threadIdx.x & 31;
  if (wid >= E + N) return;
  int src, dst;
  if (wid < E) { src = ei[wid]; dst = ei[E + wid]; }
  else         { src = wid - E; dst = src; }

  // lane covers 4 contiguous channels; all in the same head (head = lane/8)
  const float4 hs = *(const float4*)(h_src + (size_t)src * HC + lane * 4);
  const float4 hd = *(const float4*)(h_dst + (size_t)dst * HC + lane * 4);
  const float4 at = *(const float4*)(att + lane * 4);
  float s = 0.f, v;
  v = hs.x + hd.x; v = v > 0.f ? v : 0.2f * v; s = fmaf(at.x, v, s);
  v = hs.y + hd.y; v = v > 0.f ? v : 0.2f * v; s = fmaf(at.y, v, s);
  v = hs.z + hd.z; v = v > 0.f ? v : 0.2f * v; s = fmaf(at.z, v, s);
  v = hs.w + hd.w; v = v > 0.f ? v : 0.2f * v; s = fmaf(at.w, v, s);
  // reduce across the 8 lanes of this head
  s += __shfl_xor(s, 1, 32);
  s += __shfl_xor(s, 2, 32);
  s += __shfl_xor(s, 4, 32);
  if ((lane & 7) == 0) {
    int h = lane >> 3;
    logits[(size_t)wid * HEADS + h] = s;
    atomicMax(maxb + (size_t)dst * HEADS + h, f2ord(s));
  }
}

// ---------- E2: exp(logit - max) + segment sum ----------
__global__ void k_edge_exp(const int* __restrict__ ei, int E, int N,
                           float* __restrict__ exlog,   // in: logit, out: ex
                           const unsigned* __restrict__ maxb,
                           float* __restrict__ denom) {
  int t = blockIdx.x * blockDim.x + threadIdx.x;
  if (t >= (E + N) * HEADS) return;
  int e = t >> 2;          // HEADS == 4
  int h = t & 3;
  int dst = (e < E) ? ei[E + e] : (e - E);
  float mx = ord2f(maxb[(size_t)dst * HEADS + h]);
  float ex = __expf(exlog[t] - mx);
  exlog[t] = ex;
  atomicAdd(denom + (size_t)dst * HEADS + h, ex);
}

// ---------- E3: weighted scatter-sum (one wave per edge, f32 atomics at L2)
__global__ void __launch_bounds__(256) k_edge_aggr(
    const int* __restrict__ ei, int E, int N,
    const float* __restrict__ h_src, const float* __restrict__ exlog,
    const float* __restrict__ denom, float* __restrict__ out) {
  int wid  = (blockIdx.x * blockDim.x + threadIdx.x) >> 5;
  int lane = threadIdx.x & 31;
  if (wid >= E + N) return;
  int src, dst;
  if (wid < E) { src = ei[wid]; dst = ei[E + wid]; }
  else         { src = wid - E; dst = src; }
  int h = lane >> 3;
  float alpha = exlog[(size_t)wid * HEADS + h] / denom[(size_t)dst * HEADS + h];
  float4 hs = *(const float4*)(h_src + (size_t)src * HC + lane * 4);
  float* o = out + (size_t)dst * HC + lane * 4;
  atomicAdd(o + 0, alpha * hs.x);
  atomicAdd(o + 1, alpha * hs.y);
  atomicAdd(o + 2, alpha * hs.z);
  atomicAdd(o + 3, alpha * hs.w);
}

// ---------- BN stats: stride is a multiple of 128 so each thread stays on
// one channel -> fully coalesced; 2 atomics per thread.
__global__ void k_bn_stats(const float* __restrict__ out, float* __restrict__ sums, int N) {
  int gid = blockIdx.x * blockDim.x + threadIdx.x;
  int stride = gridDim.x * blockDim.x;   // launch config keeps this % 128 == 0
  int ch = gid & 127;
  float s = 0.f, sq = 0.f;
  for (int t = gid; t < N * HC; t += stride) {
    float v = out[t];
    s += v;
    sq = fmaf(v, v, sq);
  }
  atomicAdd(sums + ch, s);
  atomicAdd(sums + 128 + ch, sq);
}

__global__ void k_bn_finalize(const float* __restrict__ sums, float* __restrict__ mrstd, int N) {
  int ch = threadIdx.x;  // 128 threads
  float inv = 1.f / (float)N;
  float mean = sums[ch] * inv;
  float var  = sums[128 + ch] * inv - mean * mean;
  mrstd[ch]       = mean;
  mrstd[128 + ch] = rsqrtf(var + 1e-5f);
}

// Note: `bias` cancels exactly through training-mode batch-norm
// ((o+b) - mean(o+b) == o - mean(o)), so it is omitted.
__global__ void k_bn_apply(float* __restrict__ out, const float* __restrict__ mrstd,
                           const float* __restrict__ gamma, const float* __restrict__ beta, int n) {
  int t = blockIdx.x * blockDim.x + threadIdx.x;
  if (t >= n) return;
  int ch = t & 127;
  float y = gamma[ch] * (out[t] - mrstd[ch]) * mrstd[128 + ch] + beta[ch];
  out[t] = y > 0.f ? y : 0.02f * y;
}

extern "C" void kernel_launch(void* const* d_in, const int* in_sizes, int n_in,
                              void* d_out, int out_size, void* d_ws, size_t ws_size,
                              hipStream_t stream) {
  const float* x     = (const float*)d_in[0];
  const int*   ei    = (const int*)d_in[1];
  const float* Ws    = (const float*)d_in[2];
  const float* Wd    = (const float*)d_in[3];
  const float* att   = (const float*)d_in[4];
  // d_in[5] bias: cancels through batch-norm, unused
  const float* gamma = (const float*)d_in[6];
  const float* beta  = (const float*)d_in[7];

  const int N  = in_sizes[0] / IN_CH;
  const int E  = in_sizes[1] / 2;
  const int ET = E + N;
  float* out = (float*)d_out;

  // workspace carve-out (~158 MB)
  char* ws = (char*)d_ws;
  size_t off = 0;
  auto carve = [&](size_t bytes) {
    void* p = ws + off;
    off = (off + bytes + 255) & ~(size_t)255;
    return p;
  };
  _Float16* xh    = (_Float16*)carve((size_t)N * IN_CH * 2);
  _Float16* wt    = (_Float16*)carve((size_t)2 * IN_CH * HC * 2);
  float*    h_src = (float*)carve((size_t)N * HC * 4);
  float*    h_dst = (float*)carve((size_t)N * HC * 4);
  float*    exlog = (float*)carve((size_t)ET * HEADS * 4);
  unsigned* maxb  = (unsigned*)carve((size_t)N * HEADS * 4);
  float*    denom = (float*)carve((size_t)N * HEADS * 4);
  float*    sums  = (float*)carve(512 * 4);
  float*    mrstd = sums + 256;

  const int tiles  = (N + 15) / 16;
  const int gblocks = (tiles + GEMM_WAVES - 1) / GEMM_WAVES;

  k_init<<<1024, 256, 0, stream>>>(out, maxb, denom, sums, N * HC, N * HEADS, 256);
  k_cvt_x<<<(N * IN_CH + 255) / 256, 256, 0, stream>>>(x, xh, N * IN_CH);
  k_cvt_w<<<(2 * IN_CH * HC + 255) / 256, 256, 0, stream>>>(Ws, Wd, wt);
  k_gemm<<<gblocks, 256, 0, stream>>>(xh, wt, h_src, h_dst, N);
  k_edge_logits<<<((size_t)ET * 32 + 255) / 256, 256, 0, stream>>>(ei, E, N, h_src, h_dst, att, exlog, maxb);
  k_edge_exp<<<((size_t)ET * HEADS + 255) / 256, 256, 0, stream>>>(ei, E, N, exlog, maxb, denom);
  k_edge_aggr<<<((size_t)ET * 32 + 255) / 256, 256, 0, stream>>>(ei, E, N, h_src, exlog, denom, out);
  k_bn_stats<<<400, 256, 0, stream>>>(out, sums, N);          // stride 102400 % 128 == 0
  k_bn_finalize<<<1, 128, 0, stream>>>(sums, mrstd, N);
  k_bn_apply<<<(N * HC + 255) / 256, 256, 0, stream>>>(out, mrstd, gamma, beta, N * HC);
}
